// MultiAttention_6339371729052
// MI455X (gfx1250) — matrix operand order repaired and checked by reading, not compiled, not run on hardware
//
#include <hip/hip_runtime.h>

// Problem constants (match reference)
#define B_    8
#define HQ_   32
#define TQ_   128
#define D_    128
#define HKV_  8
#define MAXB_ 64
#define BS_   128
#define NB_   32
#define REP_  4
#define SCALE_ 0.08838834764831845f   // 128^-0.5

#define KSTR 136    // padded bf16 row stride for K tile in LDS
#define VSTR 136    // padded bf16 row stride for V^T tile in LDS
#define PSTR 128    // row stride for per-wave P scratch

typedef __attribute__((ext_vector_type(16))) __bf16 v16bf;
typedef __attribute__((ext_vector_type(2)))  __bf16 v2bf;
typedef __attribute__((ext_vector_type(8)))  float  v8f;

// Native f32 -> bf16 (fptrunc; selects hardware cvt on gfx1250)
static __device__ __forceinline__ __bf16 tobf(float f) {
  return (__bf16)f;
}
static __device__ __forceinline__ unsigned cvt2(float a, float b) {
  v2bf p;
  p.x = (__bf16)a;
  p.y = (__bf16)b;
  return __builtin_bit_cast(unsigned, p);
}

// B-matrix fragment: runs [p..p+7] and [p+16..p+23] bf16 (two ds_load_b128)
static __device__ __forceinline__ v16bf ldsB16(const __bf16* p) {
  union { uint4 u[2]; v16bf v; } r;
  r.u[0] = *(const uint4*)(p);
  r.u[1] = *(const uint4*)(p + 16);
  return r.v;
}
// A-matrix fragment from LDS: runs [k0..k0+7] and [k0+16..k0+23]
static __device__ __forceinline__ v16bf ldsA16(const __bf16* rowp, int k0) {
  union { uint4 u[2]; v16bf v; } r;
  r.u[0] = *(const uint4*)(rowp + k0);
  r.u[1] = *(const uint4*)(rowp + k0 + 16);
  return r.v;
}
// A-matrix fragment from global fp32, scaled and converted
static __device__ __forceinline__ v16bf gA(const float* rowp, int k0, float scale) {
  union { unsigned u[8]; v16bf v; } r;
  float4 f0 = *(const float4*)(rowp + k0);
  float4 f1 = *(const float4*)(rowp + k0 + 4);
  float4 f2 = *(const float4*)(rowp + k0 + 16);
  float4 f3 = *(const float4*)(rowp + k0 + 20);
  r.u[0] = cvt2(f0.x * scale, f0.y * scale);
  r.u[1] = cvt2(f0.z * scale, f0.w * scale);
  r.u[2] = cvt2(f1.x * scale, f1.y * scale);
  r.u[3] = cvt2(f1.z * scale, f1.w * scale);
  r.u[4] = cvt2(f2.x * scale, f2.y * scale);
  r.u[5] = cvt2(f2.z * scale, f2.w * scale);
  r.u[6] = cvt2(f3.x * scale, f3.y * scale);
  r.u[7] = cvt2(f3.z * scale, f3.w * scale);
  return r.v;
}

// One KV block of flash attention. MASKED=true only for the partial tail block.
template <bool MASKED>
static __device__ __forceinline__ void process_block(
    int j, int ctx,
    const __bf16* __restrict__ kl, const __bf16* __restrict__ vtl,
    __bf16* __restrict__ pw,
    const v16bf* qa, v8f* o, float* mrow, float* lrow,
    int ln, int half) {
  // ---- S = Q K^T : 8 N-tiles x 4 K-chunks of WMMA (Q pre-scaled) ----
  v8f s[8];
  #pragma unroll
  for (int n = 0; n < 8; ++n) {
    v8f acc = (v8f){};
    const __bf16* kb = kl + (n * 16 + ln) * KSTR + 8 * half;
    #pragma unroll
    for (int c = 0; c < 4; ++c) {
      v16bf bm = ldsB16(kb + c * 32);
      acc = __builtin_amdgcn_wmma_f32_16x16x32_bf16(
                false, qa[c], false, bm, (short)0, acc, false, false);
    }
    s[n] = acc;
  }

  // ---- (mask +) tile row-max ----
  float tm[8];
  #pragma unroll
  for (int i = 0; i < 8; ++i) tm[i] = -3.0e38f;
  #pragma unroll
  for (int n = 0; n < 8; ++n) {
    if (MASKED) {
      const bool valid = (j * BS_ + n * 16 + ln) < ctx;
      #pragma unroll
      for (int i = 0; i < 8; ++i) {
        float sv2 = valid ? s[n][i] : -3.0e38f;
        s[n][i] = sv2;
        tm[i] = fmaxf(tm[i], sv2);
      }
    } else {
      #pragma unroll
      for (int i = 0; i < 8; ++i) tm[i] = fmaxf(tm[i], s[n][i]);
    }
  }
  #pragma unroll
  for (int mk = 1; mk < 16; mk <<= 1) {
    #pragma unroll
    for (int i = 0; i < 8; ++i)
      tm[i] = fmaxf(tm[i], __shfl_xor(tm[i], mk, 32));
  }

  // ---- online softmax update ----
  float mn[8], sf[8];
  #pragma unroll
  for (int i = 0; i < 8; ++i) {
    mn[i] = fmaxf(mrow[i], tm[i]);
    sf[i] = __expf(mrow[i] - mn[i]);
    mrow[i] = mn[i];
    lrow[i] *= sf[i];
  }
  #pragma unroll
  for (int n = 0; n < 8; ++n) {
    #pragma unroll
    for (int i = 0; i < 8; ++i) o[n][i] *= sf[i];
  }
  float rs[8];
  #pragma unroll
  for (int i = 0; i < 8; ++i) rs[i] = 0.0f;
  #pragma unroll
  for (int n = 0; n < 8; ++n) {
    #pragma unroll
    for (int i = 0; i < 8; ++i) {
      float p = __expf(s[n][i] - mn[i]);
      s[n][i] = p;
      rs[i] += p;
    }
  }
  #pragma unroll
  for (int mk = 1; mk < 16; mk <<= 1) {
    #pragma unroll
    for (int i = 0; i < 8; ++i) rs[i] += __shfl_xor(rs[i], mk, 32);
  }
  #pragma unroll
  for (int i = 0; i < 8; ++i) lrow[i] += rs[i];

  // ---- spill P (C/D layout) to per-wave LDS as bf16, reload in A layout ----
  #pragma unroll
  for (int n = 0; n < 8; ++n) {
    #pragma unroll
    for (int i = 0; i < 8; ++i)
      pw[(i + 8 * half) * PSTR + n * 16 + ln] = tobf(s[n][i]);
  }
  // LDS ops are in-order within a wave: reload is safe without a barrier.
  v16bf pa[4];
  #pragma unroll
  for (int tc = 0; tc < 4; ++tc)
    pa[tc] = ldsA16(pw + ln * PSTR, tc * 32 + 8 * half);

  // ---- O += P V : 8 d-tiles x 4 token-chunks of WMMA ----
  #pragma unroll
  for (int n = 0; n < 8; ++n) {
    const __bf16* vb = vtl + (n * 16 + ln) * VSTR + 8 * half;
    v8f acc = o[n];
    #pragma unroll
    for (int tc = 0; tc < 4; ++tc) {
      v16bf bm = ldsB16(vb + tc * 32);
      acc = __builtin_amdgcn_wmma_f32_16x16x32_bf16(
                false, pa[tc], false, bm, (short)0, acc, false, false);
    }
    o[n] = acc;
  }
}

__global__ __launch_bounds__(256)
void paged_attn_fa_kernel(const float* __restrict__ q,
                          const float* __restrict__ sk,
                          const float* __restrict__ sv,
                          const int*   __restrict__ bt,
                          const int*   __restrict__ ctxp,
                          float*       __restrict__ out) {
  extern __shared__ __bf16 smem[];
  __bf16* kl  = smem;                      // [128][KSTR]  K tile (row-major, bf16)
  __bf16* vtl = smem + 128 * KSTR;         // [128][VSTR]  V^T tile (d-major, bf16)
  __bf16* pl  = vtl  + 128 * VSTR;         // [8 waves][16][PSTR] P scratch

  const int bh   = blockIdx.x;             // 0..63 = b*HKV + h
  const int b    = bh / HKV_;
  const int h    = bh % HKV_;
  const int tid  = threadIdx.x;
  const int wave = tid >> 5;
  const int lane = tid & 31;
  const int ln   = lane & 15;
  const int half = lane >> 4;

  const int ctx = ctxp[b];

  const int mt = blockIdx.y * 8 + wave;    // M-tile 0..31 over the 512 rows
  const int g0 = mt * 16;
  const int r  = g0 >> 7;                  // GQA rep index (uniform per tile)
  const int hq = h * REP_ + r;

  // Resident bf16 Q A-fragments, pre-scaled by 1/sqrt(D)
  const int    t_lane = (g0 + ln) & 127;
  const float* qrow   = q + (((size_t)b * HQ_ + hq) * TQ_ + t_lane) * D_;
  v16bf qa[4];
  #pragma unroll
  for (int c = 0; c < 4; ++c) qa[c] = gA(qrow, c * 32 + 8 * half, SCALE_);

  // Flash accumulators: O (16x128 f32), per-row running max/sum
  v8f o[8];
  #pragma unroll
  for (int n = 0; n < 8; ++n) o[n] = (v8f){};
  float mrow[8], lrow[8];
  #pragma unroll
  for (int i = 0; i < 8; ++i) { mrow[i] = -3.0e38f; lrow[i] = 0.0f; }

  __bf16* pw = pl + wave * (16 * PSTR);

  const size_t kvhead = ((size_t)(b * HKV_ + h)) * MAXB_;

  const int nfull = ctx >> 7;              // fully-valid KV blocks
  const int nblk  = (ctx + BS_ - 1) >> 7;  // total KV blocks (incl. partial tail)

  for (int j = 0; j < nblk; ++j) {
    const int    phys = bt[b * NB_ + j];
    const float* kgb  = sk + (kvhead + phys) * (size_t)(BS_ * D_);
    const float* vgb  = sv + (kvhead + phys) * (size_t)(BS_ * D_);

    // Prefetch next physical block into L2 (global_prefetch_b8)
    if (j + 1 < nblk) {
      const int pn = bt[b * NB_ + j + 1];
      const float* nk = sk + (kvhead + pn) * (size_t)(BS_ * D_);
      const float* nv = sv + (kvhead + pn) * (size_t)(BS_ * D_);
      __builtin_prefetch(nk + tid * 64, 0, 1);
      __builtin_prefetch(nv + tid * 64, 0, 1);
    }

    __syncthreads();   // previous tile's consumers done before overwrite
    // Cooperative load: K -> bf16 row-major, V -> bf16 transposed
    #pragma unroll
    for (int it = 0; it < 16; ++it) {
      const int vec  = tid + it * 256;     // 0..4095 float4 vectors
      const int rowi = vec >> 5;           // token 0..127
      const int c4   = (vec & 31) << 2;    // d 0..124
      float4 kf = *(const float4*)(kgb + rowi * D_ + c4);
      float4 vf = *(const float4*)(vgb + rowi * D_ + c4);
      *(uint2*)&kl[rowi * KSTR + c4] = make_uint2(cvt2(kf.x, kf.y), cvt2(kf.z, kf.w));
      vtl[(c4 + 0) * VSTR + rowi] = tobf(vf.x);
      vtl[(c4 + 1) * VSTR + rowi] = tobf(vf.y);
      vtl[(c4 + 2) * VSTR + rowi] = tobf(vf.z);
      vtl[(c4 + 3) * VSTR + rowi] = tobf(vf.w);
    }
    __syncthreads();

    if (j < nfull) {
      process_block<false>(j, ctx, kl, vtl, pw, qa, o, mrow, lrow, ln, half);
    } else {
      process_block<true>(j, ctx, kl, vtl, pw, qa, o, mrow, lrow, ln, half);
    }
  }

  // ---- epilogue: normalize and store ----
  float inv[8];
  #pragma unroll
  for (int i = 0; i < 8; ++i) inv[i] = 1.0f / fmaxf(lrow[i], 1e-9f);

  float* ob = out + (((size_t)b * HQ_ + hq) * TQ_) * (size_t)D_;
  #pragma unroll
  for (int n = 0; n < 8; ++n) {
    #pragma unroll
    for (int i = 0; i < 8; ++i) {
      const int t = (g0 + i + 8 * half) & 127;
      ob[(size_t)t * D_ + n * 16 + ln] = o[n][i] * inv[i];
    }
  }
}

extern "C" void kernel_launch(void* const* d_in, const int* in_sizes, int n_in,
                              void* d_out, int out_size, void* d_ws, size_t ws_size,
                              hipStream_t stream) {
  (void)in_sizes; (void)n_in; (void)out_size; (void)d_ws; (void)ws_size;
  const float* q   = (const float*)d_in[0];
  const float* sk  = (const float*)d_in[1];
  const float* sv  = (const float*)d_in[2];
  const int*   bt  = (const int*)d_in[3];
  const int*   ctx = (const int*)d_in[4];
  float*       out = (float*)d_out;

  dim3 grid(B_ * HKV_, (REP_ * TQ_) / 128);    // (64, 4)
  dim3 block(256);
  size_t shmem = (size_t)(128 * KSTR + 128 * VSTR + 8 * 16 * PSTR) * sizeof(__bf16);
  paged_attn_fa_kernel<<<grid, block, shmem, stream>>>(q, sk, sv, bt, ctx, out);
}
